// LSTMAttentionDot_66391604461811
// MI455X (gfx1250) — matrix operand order repaired and checked by reading, not compiled, not run on hardware
//
#include <hip/hip_runtime.h>

// ---------------------------------------------------------------------------
// LSTM + dot attention, MI455X (gfx1250, wave32).
// GEMMs use v_wmma_f32_16x16x32_bf16 (bf16 in, f32 accumulate).
// All matrix operands pre-packed into exact WMMA lane layouts; GEMM inner
// loop is a ping-pong software pipeline (2 operand register sets) with
// sched_barrier fences so loads stay one stage ahead of the WMMAs.
// Sizes: T=256, B=128, S=256, H=I=512.
// ---------------------------------------------------------------------------

typedef __attribute__((ext_vector_type(16))) __bf16          v16bf;
typedef __attribute__((ext_vector_type(16))) unsigned short  v16us;
typedef __attribute__((ext_vector_type(8)))  float           v8f;

#define T_STEPS 256
#define BATCH   128
#define SEQ     256
#define HID     512
#define INP     512
#define G4H     2048   // 4*HID
#define KCAT    1024   // INP+HID (gates K) == 2*HID (h_tilde K)

// f32 -> bf16 RNE (matches WMMA rounding)
__device__ __forceinline__ unsigned short f2bfu(float f) {
  unsigned u = __builtin_bit_cast(unsigned, f);
  u += 0x7FFFu + ((u >> 16) & 1u);
  return (unsigned short)(u >> 16);
}

// ---------------------------------------------------------------------------
// A-operand pack layout for an activation matrix Act[128][512] (M x K):
// tiles (mTile 0..7, kb 0..15), tile = mTile*16+kb, 512 bf16 per tile,
// within tile: lane*16 + e, where (ISA 16-bit A layout)
//   M = mTile*16 + (lane&15),  K = kb*32 + e + 8*(lane>>4) + 8*(e>>3)
// ---------------------------------------------------------------------------
__device__ __forceinline__ size_t apack_idx(int row, int colK) {
  int c    = colK & 31;
  int half = (c >> 3) & 1;
  int e    = (c & 7) + 8 * ((c >> 4) & 1);
  int lane = (row & 15) | (half << 4);
  int tile = (row >> 4) * 16 + (colK >> 5);
  return ((size_t)tile << 9) + (size_t)(lane * 16 + e);
}

// Pack nMat consecutive [128][512] f32 matrices into A layout.
__global__ void pack_a_multi(const float* __restrict__ src,
                             unsigned short* __restrict__ dst) {
  int gid  = blockIdx.x * 256 + threadIdx.x;
  int mat  = gid >> 16;
  int rem  = gid & 65535;
  int tile = rem >> 9;              // 0..127
  int lane = (rem >> 4) & 31;
  int e    = rem & 15;
  int mTile = tile >> 4, kb = tile & 15;
  int M = mTile * 16 + (lane & 15);
  int K = kb * 32 + e + 8 * (lane >> 4) + 8 * (e >> 3);
  dst[gid] = f2bfu(src[(size_t)mat * (BATCH * 512) + (size_t)M * 512 + K]);
}

// ---------------------------------------------------------------------------
// B-operand pack: tile (32x16 bf16): lane L holds column N=L&15,
// K = (e&7) + 8*(L>>4) + 16*(e>>3). Tiles contiguous, 512 bf16 each.
// ---------------------------------------------------------------------------
__global__ void pack_w_gates(const float* __restrict__ Wi,
                             const float* __restrict__ Wh,
                             unsigned short* __restrict__ wp) {
  int gid  = blockIdx.x * 256 + threadIdx.x;     // < 1024*2048
  int e    = gid & 15;
  int lane = (gid >> 4) & 31;
  int tile = gid >> 9;
  int kb   = tile >> 7;
  int nb   = tile & 127;
  int K    = kb * 32 + (e & 7) + ((lane >> 4) << 3) + ((e >> 3) << 4);
  int N    = nb * 16 + (lane & 15);
  float v  = (K < INP) ? Wi[(size_t)N * INP + K]
                       : Wh[(size_t)N * HID + (K - INP)];
  wp[gid] = f2bfu(v);
}

__global__ void pack_w_ctx(const float* __restrict__ Wc,
                           unsigned short* __restrict__ wp) {
  int gid  = blockIdx.x * 256 + threadIdx.x;     // < 1024*512
  int e    = gid & 15;
  int lane = (gid >> 4) & 31;
  int tile = gid >> 9;
  int kb   = tile >> 5;
  int nb   = tile & 31;
  int K    = kb * 32 + (e & 7) + ((lane >> 4) << 3) + ((e >> 3) << 4);
  int N    = nb * 16 + (lane & 15);
  wp[gid] = f2bfu(Wc[(size_t)N * KCAT + K]);
}

__global__ void bias_sum_k(const float* __restrict__ bi,
                           const float* __restrict__ bh,
                           float* __restrict__ bs) {
  int n = blockIdx.x * 256 + threadIdx.x;
  bs[n] = bi[n] + bh[n];
}

// ---------------------------------------------------------------------------
// GEMM helpers
// ---------------------------------------------------------------------------
__device__ __forceinline__ v16bf ld_atile(const unsigned short* __restrict__ lo,
                                          const unsigned short* __restrict__ hi,
                                          int mTile, int kb, int lofs) {
  const unsigned short* p = (kb < 16)
      ? (lo + (((size_t)(mTile * 16 + kb)) << 9) + lofs)
      : (hi + (((size_t)(mTile * 16 + (kb - 16))) << 9) + lofs);
  return *(const v16bf*)p;
}

#define WMMA_BF16(A, B, C) \
  __builtin_amdgcn_wmma_f32_16x16x32_bf16(false, (A), false, (B), (short)0, (C), false, false)

// Ping-pong pipelined accumulation over 32 K-blocks.
// lo/hi: A-pack sources (kb<16 / kb>=16). wp: B-pack, TPK tiles per K-block.
__device__ __forceinline__ void gemm_core(const unsigned short* __restrict__ lo,
                                          const unsigned short* __restrict__ hi,
                                          const unsigned short* __restrict__ wp,
                                          int mTile, int nb0, int lofs, int TPK,
                                          v8f acc[4]) {
  // prologue: set P <- kb=0
  v16bf a0 = ld_atile(lo, hi, mTile, 0, lofs);
  const unsigned short* t0 = wp + (((size_t)nb0) << 9) + lofs;
  v16bf p0 = *(const v16bf*)(t0);
  v16bf p1 = *(const v16bf*)(t0 + 512);
  v16bf p2 = *(const v16bf*)(t0 + 1024);
  v16bf p3 = *(const v16bf*)(t0 + 1536);

  for (int kb = 0; kb < 32; kb += 2) {
    // stage 1: set Q <- kb+1
    v16bf a1 = ld_atile(lo, hi, mTile, kb + 1, lofs);
    const unsigned short* t1 = wp + (((size_t)((kb + 1) * TPK + nb0)) << 9) + lofs;
    v16bf q0 = *(const v16bf*)(t1);
    v16bf q1 = *(const v16bf*)(t1 + 512);
    v16bf q2 = *(const v16bf*)(t1 + 1024);
    v16bf q3 = *(const v16bf*)(t1 + 1536);
    __builtin_amdgcn_sched_barrier(0);
    // stage 2: consume set P (kb) while Q loads are in flight
    acc[0] = WMMA_BF16(a0, p0, acc[0]);
    acc[1] = WMMA_BF16(a0, p1, acc[1]);
    acc[2] = WMMA_BF16(a0, p2, acc[2]);
    acc[3] = WMMA_BF16(a0, p3, acc[3]);
    __builtin_amdgcn_sched_barrier(0);
    // stage 3: set P <- kb+2 (clamped; last refill is redundant, harmless)
    const int n2 = (kb + 2 < 32) ? kb + 2 : 31;
    a0 = ld_atile(lo, hi, mTile, n2, lofs);
    const unsigned short* t2 = wp + (((size_t)(n2 * TPK + nb0)) << 9) + lofs;
    p0 = *(const v16bf*)(t2);
    p1 = *(const v16bf*)(t2 + 512);
    p2 = *(const v16bf*)(t2 + 1024);
    p3 = *(const v16bf*)(t2 + 1536);
    __builtin_amdgcn_sched_barrier(0);
    // stage 4: consume set Q (kb+1) while P loads are in flight
    acc[0] = WMMA_BF16(a1, q0, acc[0]);
    acc[1] = WMMA_BF16(a1, q1, acc[1]);
    acc[2] = WMMA_BF16(a1, q2, acc[2]);
    acc[3] = WMMA_BF16(a1, q3, acc[3]);
    __builtin_amdgcn_sched_barrier(0);
  }
}

// ---------------------------------------------------------------------------
// Gates GEMM: gates[128,2048] = [x_t | h] (128x1024) @ W2 (1024x2048) + bias.
// 256 waves, each owns a 16x64 strip.
// ---------------------------------------------------------------------------
__global__ void gemm_gates(const unsigned short* __restrict__ xap,
                           const unsigned short* __restrict__ hap,
                           const unsigned short* __restrict__ wp,
                           const float* __restrict__ bias,
                           float* __restrict__ gates) {
  const int lane   = threadIdx.x & 31;
  const int half   = lane >> 4;
  const int W      = blockIdx.x * 8 + (threadIdx.x >> 5);  // 0..255
  const int mTile  = W >> 5;                                // 0..7
  const int nStrip = W & 31;                                // 0..31
  const int lofs   = lane * 16;

  v8f acc[4];
#pragma unroll
  for (int j = 0; j < 4; ++j) {
    float bv = bias[nStrip * 64 + j * 16 + (lane & 15)];
#pragma unroll
    for (int r = 0; r < 8; ++r) acc[j][r] = bv;
  }

  gemm_core(xap, hap, wp, mTile, nStrip * 4, lofs, 128, acc);

#pragma unroll
  for (int j = 0; j < 4; ++j) {
    int col = nStrip * 64 + j * 16 + (lane & 15);
#pragma unroll
    for (int r = 0; r < 8; ++r) {
      int row = mTile * 16 + r + 8 * half;
      gates[(size_t)row * G4H + col] = acc[j][r];
    }
  }
}

// ---------------------------------------------------------------------------
// LSTM pointwise; also emits hy in bf16 A-pack layout for the h_tilde GEMM.
// ---------------------------------------------------------------------------
__global__ void lstm_pw(const float* __restrict__ gates,
                        float* __restrict__ c,
                        float* __restrict__ hy,
                        unsigned short* __restrict__ hyp) {
  int idx = blockIdx.x * 256 + threadIdx.x;      // < 65536
  int b = idx >> 9, h = idx & 511;
  const float* g = gates + (size_t)b * G4H;
  float gi = g[h], gf = g[HID + h], gg = g[2 * HID + h], go = g[3 * HID + h];
  float si = 1.f / (1.f + __expf(-gi));
  float sf = 1.f / (1.f + __expf(-gf));
  float so = 1.f / (1.f + __expf(-go));
  float cy = sf * c[idx] + si * tanhf(gg);
  float hv = so * tanhf(cy);
  c[idx]  = cy;
  hy[idx] = hv;
  hyp[apack_idx(b, h)] = f2bfu(hv);
}

// ---------------------------------------------------------------------------
// Attention (per batch row): raw-view scores, softmax, weighted context.
// ctx (67MB f32) is L2-resident (192MB L2) -> L2-bandwidth VALU work.
// Emits wc directly in bf16 A-pack layout.
// ---------------------------------------------------------------------------
__global__ void attn_k(const float* __restrict__ ctx,
                       const float* __restrict__ hy,
                       unsigned short* __restrict__ wcp) {
  const int b   = blockIdx.x;     // 0..127
  const int tid = threadIdx.x;    // 0..255
  __shared__ float sh_h[HID];
  __shared__ float sh_a[SEQ];
  __shared__ float red[256];

  sh_h[tid]       = hy[(size_t)b * HID + tid];
  sh_h[tid + 256] = hy[(size_t)b * HID + tid + 256];
  __syncthreads();

  const float* cr = ctx + (size_t)b * (HID * SEQ) + tid;
  float sc = 0.f;
#pragma unroll 4
  for (int h = 0; h < HID; ++h) sc = fmaf(sh_h[h], cr[(size_t)h * SEQ], sc);

  red[tid] = sc; __syncthreads();
  for (int off = 128; off > 0; off >>= 1) {
    if (tid < off) red[tid] = fmaxf(red[tid], red[tid + off]);
    __syncthreads();
  }
  float mx = red[0]; __syncthreads();
  float e = __expf(sc - mx);
  red[tid] = e; __syncthreads();
  for (int off = 128; off > 0; off >>= 1) {
    if (tid < off) red[tid] += red[tid + off];
    __syncthreads();
  }
  sh_a[tid] = e * (1.0f / red[0]);
  __syncthreads();

  float a0 = 0.f, a1 = 0.f;
  const float* cp = ctx + (size_t)b * HID;
  for (int s = 0; s < SEQ; ++s) {
    // pull the next S-row toward the WGP while current FMAs run
    int sp = (s + 4 < SEQ) ? s + 4 : SEQ - 1;
    __builtin_prefetch(cp + (size_t)sp * (BATCH * HID) + tid, 0, 1);
    float al = sh_a[s];
    const float* row = cp + (size_t)s * (BATCH * HID);
    a0 = fmaf(al, row[tid], a0);
    a1 = fmaf(al, row[tid + 256], a1);
  }
  wcp[apack_idx(b, tid)]       = f2bfu(a0);
  wcp[apack_idx(b, tid + 256)] = f2bfu(a1);
}

// ---------------------------------------------------------------------------
// h_tilde GEMM: h = tanh([wc | hy] (128x1024) @ Wc^T (1024x512)).
// Writes out[t] (f32) and the new carry directly in bf16 A-pack layout.
// ---------------------------------------------------------------------------
__global__ void gemm_htilde(const unsigned short* __restrict__ wcp,
                            const unsigned short* __restrict__ hyp,
                            const unsigned short* __restrict__ wp2,
                            unsigned short* __restrict__ hpack,
                            float* __restrict__ outp, int t) {
  const int lane   = threadIdx.x & 31;
  const int half   = lane >> 4;
  const int W      = blockIdx.x * 8 + (threadIdx.x >> 5);  // 0..63
  const int mTile  = W >> 3;                                // 0..7
  const int nStrip = W & 7;                                 // 0..7
  const int lofs   = lane * 16;

  v8f acc[4];
#pragma unroll
  for (int j = 0; j < 4; ++j)
#pragma unroll
    for (int r = 0; r < 8; ++r) acc[j][r] = 0.f;

  gemm_core(wcp, hyp, wp2, mTile, nStrip * 4, lofs, 32, acc);

  float* outT = outp + (size_t)t * (BATCH * HID);
#pragma unroll
  for (int j = 0; j < 4; ++j) {
    int col = nStrip * 64 + j * 16 + (lane & 15);
#pragma unroll
    for (int r = 0; r < 8; ++r) {
      int row = mTile * 16 + r + 8 * half;
      float v = tanhf(acc[j][r]);
      outT[(size_t)row * HID + col] = v;
      hpack[apack_idx(row, col)]    = f2bfu(v);
    }
  }
}

// ---------------------------------------------------------------------------
// Launch
// ---------------------------------------------------------------------------
extern "C" void kernel_launch(void* const* d_in, const int* in_sizes, int n_in,
                              void* d_out, int out_size, void* d_ws, size_t ws_size,
                              hipStream_t stream) {
  (void)in_sizes; (void)n_in; (void)out_size;
  const float* x   = (const float*)d_in[0];
  const float* h0  = (const float*)d_in[1];
  const float* c0  = (const float*)d_in[2];
  const float* ctx = (const float*)d_in[3];
  const float* Wi  = (const float*)d_in[4];
  const float* Wh  = (const float*)d_in[5];
  const float* bi  = (const float*)d_in[6];
  const float* bh  = (const float*)d_in[7];
  const float* Wc  = (const float*)d_in[8];
  float* out = (float*)d_out;

  // Workspace carve-up (256B-aligned)
  char*  ws  = (char*)d_ws;
  size_t off = 0;
  auto carve = [&](size_t bytes) -> char* {
    char* p = ws + off;
    off = (off + bytes + 255) & ~(size_t)255;
    return p;
  };
  unsigned short* wp1    = (unsigned short*)carve((size_t)KCAT * G4H * 2); // 4 MB
  unsigned short* wp2    = (unsigned short*)carve((size_t)KCAT * HID * 2); // 1 MB
  float*          bias   = (float*)carve(G4H * 4);
  float*          gates  = (float*)carve((size_t)BATCH * G4H * 4);         // 1 MB
  float*          cbuf   = (float*)carve((size_t)BATCH * HID * 4);
  float*          hybuf  = (float*)carve((size_t)BATCH * HID * 4);
  unsigned short* hpack  = (unsigned short*)carve((size_t)BATCH * HID * 2);
  unsigned short* hypack = (unsigned short*)carve((size_t)BATCH * HID * 2);
  unsigned short* wcpack = (unsigned short*)carve((size_t)BATCH * HID * 2);
  unsigned short* xtp    = (unsigned short*)carve((size_t)BATCH * INP * 2);
  size_t xfull_bytes = (size_t)T_STEPS * BATCH * INP * 2;                  // 33.5 MB
  unsigned short* xpack = (unsigned short*)carve(xfull_bytes);
  const bool bigws = (ws_size >= off);   // launch-invariant -> deterministic

  // Setup
  pack_w_gates<<<8192, 256, 0, stream>>>(Wi, Wh, wp1);
  pack_w_ctx  <<<2048, 256, 0, stream>>>(Wc, wp2);
  bias_sum_k  <<<8,    256, 0, stream>>>(bi, bh, bias);
  pack_a_multi<<<256,  256, 0, stream>>>(h0, hpack);
  hipMemcpyAsync(cbuf, c0, (size_t)BATCH * HID * sizeof(float),
                 hipMemcpyDeviceToDevice, stream);
  if (bigws)
    pack_a_multi<<<T_STEPS * 256, 256, 0, stream>>>(x, xpack);

  // Serial recurrence
  for (int t = 0; t < T_STEPS; ++t) {
    const unsigned short* xap;
    if (bigws) {
      xap = xpack + (size_t)t * (BATCH * INP);
    } else {
      pack_a_multi<<<256, 256, 0, stream>>>(x + (size_t)t * (BATCH * INP), xtp);
      xap = xtp;
    }
    gemm_gates <<<32,  256, 0, stream>>>(xap, hpack, wp1, bias, gates);
    lstm_pw    <<<256, 256, 0, stream>>>(gates, cbuf, hybuf, hypack);
    attn_k     <<<128, 256, 0, stream>>>(ctx, hybuf, wcpack);
    gemm_htilde<<<8,   256, 0, stream>>>(wcpack, hypack, wp2, hpack, out, t);
  }

  // hT = out[T-1]; cT from cbuf
  const size_t BH = (size_t)BATCH * HID;
  hipMemcpyAsync(out + (size_t)T_STEPS * BH, out + (size_t)(T_STEPS - 1) * BH,
                 BH * sizeof(float), hipMemcpyDeviceToDevice, stream);
  hipMemcpyAsync(out + (size_t)T_STEPS * BH + BH, cbuf,
                 BH * sizeof(float), hipMemcpyDeviceToDevice, stream);
}
